// BahdanauAttention_32615981646521
// MI455X (gfx1250) — matrix operand order repaired and checked by reading, not compiled, not run on hardware
//
#include <hip/hip_runtime.h>
#include <hip/hip_bf16.h>
#include <math.h>

// ---------------------------------------------------------------------------
// Bahdanau attention, MI455X (gfx1250, wave32, WMMA + async-to-LDS).
// B=64, S=2048, D=512, U=512.
// ---------------------------------------------------------------------------

#define BB 64
#define SS 2048
#define DD 512
#define UU 512

typedef __attribute__((ext_vector_type(16))) __bf16  v16bf;
typedef __attribute__((ext_vector_type(8)))  float   v8f;
typedef __attribute__((ext_vector_type(8)))  unsigned int v8u;

union BFrag { v8u u; v16bf v; };
union AFrag { unsigned short u[16]; v16bf v; };

__device__ __forceinline__ unsigned short f2bf_u16(float f) {
    union { float f; unsigned u; } c; c.f = f;
    unsigned u = c.u;
    unsigned r = (u + 0x7FFFu + ((u >> 16) & 1u)) >> 16;   // RNE
    return (unsigned short)r;
}

// CDNA5 async copy: 16B per lane, global -> LDS, tracked by ASYNCcnt.
__device__ __forceinline__ void async_copy_b128(unsigned lds_off, const void* gaddr) {
    asm volatile("global_load_async_to_lds_b128 %0, %1, off"
                 :: "v"(lds_off), "v"(gaddr) : "memory");
}
__device__ __forceinline__ void wait_async_le8() {
    asm volatile("s_wait_asynccnt 0x8" ::: "memory");
}
__device__ __forceinline__ void wait_async_0() {
    asm volatile("s_wait_asynccnt 0x0" ::: "memory");
}

// ---------------------------------------------------------------------------
// Kernel 1: W2 [D,U] fp32 -> bf16, fragment-major for the WMMA B operand.
// (ks, t, l, i):  k = ks*32 + 16*(l>=16) + i ; n = t*16 + (l&15)
// flat out index = ((ks*32 + t)*32 + l)*16 + i   (32B per lane, contiguous).
// ---------------------------------------------------------------------------
__global__ __launch_bounds__(256) void w2_swizzle_kernel(
    const float* __restrict__ W2, unsigned short* __restrict__ out) {
    int idx = blockIdx.x * 256 + threadIdx.x;      // 0 .. 262143
    int i  = idx & 15;
    int l  = (idx >> 4) & 31;
    int t  = (idx >> 9) & 31;
    int ks = idx >> 14;
    int k  = (ks << 5) + ((l >> 4) << 4) + i;
    int n  = (t << 4) + (l & 15);
    out[idx] = f2bf_u16(W2[k * UU + n]);
}

// ---------------------------------------------------------------------------
// Kernel 2: qpb[b][n] = query[b] . W1[:,n] + b1[n] + b2[n]
// ---------------------------------------------------------------------------
__global__ __launch_bounds__(256) void qproj_kernel(
    const float* __restrict__ query, const float* __restrict__ W1,
    const float* __restrict__ b1, const float* __restrict__ b2,
    float* __restrict__ qpb) {
    int idx = blockIdx.x * 256 + threadIdx.x;      // 0 .. 32767
    int n = idx & (UU - 1);
    int b = idx >> 9;
    float acc = b1[n] + b2[n];
    const float* qr = query + b * DD;
    #pragma unroll 8
    for (int d = 0; d < DD; ++d)
        acc = fmaf(qr[d], W1[d * UU + n], acc);
    qpb[idx] = acc;
}

// ---------------------------------------------------------------------------
// Kernel 3: scores.  One block per (b, 32-row s-tile).  8 waves x 64 U-cols,
// 2 A-tiles per wave (8 accumulators), B double-buffered in LDS via
// global_load_async_to_lds_b128 (per-wave slices, no barrier needed for B).
// ---------------------------------------------------------------------------
__global__ __launch_bounds__(256) void scores_kernel(
    const float* __restrict__ values,
    const unsigned short* __restrict__ W2bf,
    const float* __restrict__ qpb,
    const float* __restrict__ Vw,
    const float* __restrict__ bv,
    float* __restrict__ scores) {

    const int b    = blockIdx.y;
    const int s0   = blockIdx.x << 5;              // 32 rows per block
    const int tid  = threadIdx.x;
    const int lane = tid & 31;
    const int wave = tid >> 5;                     // 0..7
    const int nbase = wave << 6;                   // 64 U-columns per wave

    __shared__ unsigned short ldsA[2][32][40];     // 32x32 bf16 (+pad), x2 buf
    __shared__ unsigned short ldsB[2][16384];      // 32KB per buffer
    __shared__ float ldsScore[32];

    const unsigned ldsB_base = (unsigned)(size_t)(void*)&ldsB[0][0];
    const size_t vrow_base = ((size_t)b * SS + s0) * DD;

    v8f acc00 = {}, acc01 = {}, acc02 = {}, acc03 = {};
    v8f acc10 = {}, acc11 = {}, acc12 = {}, acc13 = {};

    const int m   = lane & 15;
    const int hi8 = (lane >> 4) << 3;

    // ---- issue async B copy for K-step ks into buffer buf (per-wave 4KB) --
    auto issueB = [&](int ks, int buf) {
        const char* g = (const char*)W2bf + ((size_t)ks << 15) + (wave << 12) + (lane << 4);
        unsigned    l = ldsB_base + (buf << 15) + (wave << 12) + (lane << 4);
        #pragma unroll
        for (int j = 0; j < 8; ++j)
            async_copy_b128(l + j * 512, g + j * 512);
    };

    // ---- stage 32x32 fp32 A-tile as bf16 into buffer buf (1 float4/thr) ---
    auto stageA = [&](int k0, int buf) {
        int e = tid << 2;                          // 4 consecutive elements
        int row = e >> 5, col = e & 31;
        float4 f = *(const float4*)&values[vrow_base + (size_t)row * DD + k0 + col];
        unsigned lo = ((unsigned)f2bf_u16(f.y) << 16) | f2bf_u16(f.x);
        unsigned hi = ((unsigned)f2bf_u16(f.w) << 16) | f2bf_u16(f.z);
        unsigned long long p = ((unsigned long long)hi << 32) | lo;
        *(unsigned long long*)&ldsA[buf][row][col] = p;   // ds_store_b64
    };

    // ---- pipeline prologue ----
    stageA(0, 0);
    issueB(0, 0);
    __syncthreads();

    for (int ks = 0; ks < 16; ++ks) {
        const int buf = ks & 1;
        if (ks < 15) {
            issueB(ks + 1, buf ^ 1);               // async prefetch next B
            stageA((ks + 1) << 5, buf ^ 1);        // stage next A tile
            wait_async_le8();                      // current B slice resident
        } else {
            wait_async_0();
        }

        // ---- A fragments for the two 16-row tiles (16-bit A 16x32 layout) -
        AFrag a0, a1;
        #pragma unroll
        for (int i = 0; i < 16; ++i) {
            int K = (hi8 << 1) + i;
            a0.u[i] = ldsA[buf][m][K];
            a1.u[i] = ldsA[buf][16 + m][K];
        }

        // ---- 4 B fragments from LDS, 8 WMMAs (B reused across A tiles) ----
        const unsigned short* bs = &ldsB[buf][(wave << 11) + (lane << 4)];
        {
            BFrag bf; bf.u = *(const v8u*)(bs + 0 * 512);
            acc00 = __builtin_amdgcn_wmma_f32_16x16x32_bf16(false, a0.v, false, bf.v, (short)0, acc00, false, false);
            acc10 = __builtin_amdgcn_wmma_f32_16x16x32_bf16(false, a1.v, false, bf.v, (short)0, acc10, false, false);
        }
        {
            BFrag bf; bf.u = *(const v8u*)(bs + 1 * 512);
            acc01 = __builtin_amdgcn_wmma_f32_16x16x32_bf16(false, a0.v, false, bf.v, (short)0, acc01, false, false);
            acc11 = __builtin_amdgcn_wmma_f32_16x16x32_bf16(false, a1.v, false, bf.v, (short)0, acc11, false, false);
        }
        {
            BFrag bf; bf.u = *(const v8u*)(bs + 2 * 512);
            acc02 = __builtin_amdgcn_wmma_f32_16x16x32_bf16(false, a0.v, false, bf.v, (short)0, acc02, false, false);
            acc12 = __builtin_amdgcn_wmma_f32_16x16x32_bf16(false, a1.v, false, bf.v, (short)0, acc12, false, false);
        }
        {
            BFrag bf; bf.u = *(const v8u*)(bs + 3 * 512);
            acc03 = __builtin_amdgcn_wmma_f32_16x16x32_bf16(false, a0.v, false, bf.v, (short)0, acc03, false, false);
            acc13 = __builtin_amdgcn_wmma_f32_16x16x32_bf16(false, a1.v, false, bf.v, (short)0, acc13, false, false);
        }
        __syncthreads();                           // publish next A tile
    }

    // ---- epilogue: h = tanh(acc + qproj + b2), score = sum_n h * V[n] -----
    if (tid < 32) ldsScore[tid] = 0.0f;
    __syncthreads();

    const int ncol  = lane & 15;
    const int mhalf = (lane >> 4) << 3;            // C layout row = r + 8*(lane>=16)
    const float* qrow = qpb + b * UU;

    v8f* accs[2][4] = { { &acc00, &acc01, &acc02, &acc03 },
                        { &acc10, &acc11, &acc12, &acc13 } };
    #pragma unroll
    for (int a = 0; a < 2; ++a) {
        #pragma unroll
        for (int t = 0; t < 4; ++t) {
            int n = nbase + (t << 4) + ncol;
            float qb = qrow[n];
            float vn = Vw[n];
            v8f av = *accs[a][t];
            #pragma unroll
            for (int r = 0; r < 8; ++r) {
                float h = tanhf(av[r] + qb);
                atomicAdd(&ldsScore[(a << 4) + mhalf + r], h * vn);  // ds_add_f32
            }
        }
    }
    __syncthreads();
    if (tid < 32)
        scores[(size_t)b * SS + s0 + tid] = ldsScore[tid] + bv[0];
}

// ---------------------------------------------------------------------------
// Kernel 4: softmax over S per batch.  64 blocks x 256 threads, 8 vals/thread.
// ---------------------------------------------------------------------------
__global__ __launch_bounds__(256) void softmax_kernel(
    const float* __restrict__ scores, float* __restrict__ attn) {
    const int b = blockIdx.x;
    const int tid = threadIdx.x;
    __shared__ float red[256];

    const float* sr = scores + (size_t)b * SS;
    float* ar = attn + (size_t)b * SS;

    float v[8];
    float mx = -3.402823466e38f;
    #pragma unroll
    for (int j = 0; j < 8; ++j) {
        v[j] = sr[tid + (j << 8)];
        mx = fmaxf(mx, v[j]);
    }
    red[tid] = mx; __syncthreads();
    for (int off = 128; off > 0; off >>= 1) {
        if (tid < off) red[tid] = fmaxf(red[tid], red[tid + off]);
        __syncthreads();
    }
    const float M = red[0];
    __syncthreads();

    float s = 0.0f;
    #pragma unroll
    for (int j = 0; j < 8; ++j) { v[j] = __expf(v[j] - M); s += v[j]; }
    red[tid] = s; __syncthreads();
    for (int off = 128; off > 0; off >>= 1) {
        if (tid < off) red[tid] += red[tid + off];
        __syncthreads();
    }
    const float inv = 1.0f / red[0];
    #pragma unroll
    for (int j = 0; j < 8; ++j) ar[tid + (j << 8)] = v[j] * inv;
}

// ---------------------------------------------------------------------------
// Kernel 5: context[b][d] = sum_s attn[b][s] * values[b][s][d]  (HBM bound)
// ---------------------------------------------------------------------------
__global__ __launch_bounds__(256) void context_kernel(
    const float* __restrict__ attn, const float* __restrict__ values,
    float* __restrict__ out) {
    const int b = blockIdx.y;
    const int d = blockIdx.x * 256 + threadIdx.x;
    const int tid = threadIdx.x;
    __shared__ float lattn[256];

    float acc = 0.0f;
    const float* vb = values + (size_t)b * SS * DD;
    for (int sc = 0; sc < SS; sc += 256) {
        __syncthreads();
        lattn[tid] = attn[(size_t)b * SS + sc + tid];
        __syncthreads();
        #pragma unroll 8
        for (int j = 0; j < 256; ++j)
            acc = fmaf(lattn[j], vb[(size_t)(sc + j) * DD + d], acc);
    }
    out[b * DD + d] = acc;
}

// ---------------------------------------------------------------------------
// Launcher
// ---------------------------------------------------------------------------
extern "C" void kernel_launch(void* const* d_in, const int* in_sizes, int n_in,
                              void* d_out, int out_size, void* d_ws, size_t ws_size,
                              hipStream_t stream) {
    const float* query = (const float*)d_in[0];   // [B,D]
    const float* values= (const float*)d_in[1];   // [B,S,D]
    const float* W1    = (const float*)d_in[2];   // [D,U]
    const float* b1    = (const float*)d_in[3];   // [U]
    const float* W2    = (const float*)d_in[4];   // [D,U]
    const float* b2    = (const float*)d_in[5];   // [U]
    const float* Vw    = (const float*)d_in[6];   // [U,1]
    const float* bv    = (const float*)d_in[7];   // [1]
    float* out = (float*)d_out;                   // [B,D]

    char* base = (char*)d_ws;
    float*          ws_qpb    = (float*)(base);                              // 128 KB
    unsigned short* ws_w2bf   = (unsigned short*)(base + 131072);            // 512 KB
    float*          ws_scores = (float*)(base + 131072 + 524288);            // 512 KB
    float*          ws_attn   = (float*)(base + 131072 + 524288 + 524288);   // 512 KB

    w2_swizzle_kernel<<<(DD * UU) / 256, 256, 0, stream>>>(W2, ws_w2bf);
    qproj_kernel<<<(BB * UU) / 256, 256, 0, stream>>>(query, W1, b1, b2, ws_qpb);
    scores_kernel<<<dim3(SS / 32, BB), 256, 0, stream>>>(
        values, ws_w2bf, ws_qpb, Vw, bv, ws_scores);
    softmax_kernel<<<BB, 256, 0, stream>>>(ws_scores, ws_attn);
    context_kernel<<<dim3(DD / 256, BB), 256, 0, stream>>>(ws_attn, values, out);
}
